// MultiHeadAttentionLayer_64733747085884
// MI455X (gfx1250) — matrix operand order repaired and checked by reading, not compiled, not run on hardware
//
#include <hip/hip_runtime.h>
#include <hip/hip_bf16.h>
#include <math.h>

#define S_LEN 2048
#define HDIM  1024
#define NHEADS 16
#define DHEAD  64

typedef __bf16 bf16_t;
typedef __attribute__((ext_vector_type(16))) __bf16 bf16x16;
typedef __attribute__((ext_vector_type(8)))  float  f32x8;
typedef __attribute__((ext_vector_type(4)))  unsigned int u32x4;
typedef __attribute__((ext_vector_type(4)))  int i32x4;

// Probe the CDNA5 async direct-to-LDS path (ASYNCcnt); fall back to
// synchronous b128 load+store staging if the builtin isn't declared.
#if __has_builtin(__builtin_amdgcn_global_load_async_to_lds_b128) && \
    __has_builtin(__builtin_amdgcn_s_wait_asynccnt)
#define USE_ASYNC_LDS 1
#else
#define USE_ASYNC_LDS 0
#endif

#if USE_ASYNC_LDS
typedef __attribute__((address_space(1))) i32x4 g_i32x4;  // global
typedef __attribute__((address_space(3))) i32x4 l_i32x4;  // LDS
#endif

__device__ __forceinline__ void stage16B(bf16_t* lds, const bf16_t* gmem) {
#if USE_ASYNC_LDS
  __builtin_amdgcn_global_load_async_to_lds_b128((g_i32x4*)gmem, (l_i32x4*)lds,
                                                 0, 0);
#else
  *(u32x4*)lds = *(const u32x4*)gmem;
#endif
}
__device__ __forceinline__ void stage_wait() {
#if USE_ASYNC_LDS
  __builtin_amdgcn_s_wait_asynccnt(0);
#endif
}

// Fragment K-index layouts (ISA 7.12.2, wave32, 16-bit):
//  A (16x32): lane half h, dwords 0-3 -> K = h*8 + [0,8), dwords 4-7 -> K = 16+h*8+[0,8)
//  B (32x16): lane half h, dwords 0-3 -> K = h*8 + [0,8), dwords 4-7 -> K = 16+h*8+[0,8)
struct fragbits { u32x4 lo, hi; };

__device__ __forceinline__ bf16x16 frag_a(const bf16_t* row, int base, int half) {
  fragbits t;
  t.lo = *(const u32x4*)(row + base + half * 8);
  t.hi = *(const u32x4*)(row + base + 16 + half * 8);
  return __builtin_bit_cast(bf16x16, t);
}
__device__ __forceinline__ bf16x16 frag_b(const bf16_t* row, int base, int half) {
  fragbits t;
  t.lo = *(const u32x4*)(row + base + half * 8);
  t.hi = *(const u32x4*)(row + base + 16 + half * 8);
  return __builtin_bit_cast(bf16x16, t);
}

// ---------------------------------------------------------------- fp32 -> bf16
__global__ void cvt_f32_to_bf16(const float* __restrict__ src,
                                bf16_t* __restrict__ dst, int n) {
  int i = blockIdx.x * blockDim.x + threadIdx.x;
  if (i < n) dst[i] = (bf16_t)src[i];
}

// -------------------- fp32 [K][N] -> bf16 transposed [N][K] (one-time, weights)
__global__ void cvt_transpose_bf16(const float* __restrict__ src,
                                   bf16_t* __restrict__ dst, int Kd, int Nd) {
  int i = blockIdx.x * blockDim.x + threadIdx.x;
  if (i >= Kd * Nd) return;
  int k = i / Nd, n = i % Nd;                 // coalesced reads over n
  dst[(size_t)n * Kd + k] = (bf16_t)src[i];
}

// ------------------------------------------------- 64x64-tile bf16 WMMA GEMM
// C[M,N] = A[M,K](bf16) @ B[K,N] + bias[N]; B passed PRE-TRANSPOSED as Bt[N][K].
__global__ __launch_bounds__(128)
void gemm_bf16_wmma(const bf16_t* __restrict__ A, const bf16_t* __restrict__ Bt,
                    const float* __restrict__ bias, float* __restrict__ C,
                    int M, int N, int K) {
  __shared__ __attribute__((aligned(16))) bf16_t As[64][64];   // [m][k]
  __shared__ __attribute__((aligned(16))) bf16_t Bts[64][64];  // [n][k]
  const int tid = threadIdx.x, lane = tid & 31, wave = tid >> 5;
  const int bm = blockIdx.y * 64, bn = blockIdx.x * 64;
  const int m = lane & 15, half = lane >> 4;

  f32x8 acc[4] = {};
  const int ar = tid >> 1;          // staging row 0..63
  const int ac = (tid & 1) * 32;    // staging col 0 / 32

  for (int k0 = 0; k0 < K; k0 += 64) {
    const bf16_t* asrc = &A [(size_t)(bm + ar) * K + k0 + ac];
    const bf16_t* bsrc = &Bt[(size_t)(bn + ar) * K + k0 + ac];
#pragma unroll
    for (int u = 0; u < 4; ++u) {
      stage16B(&As [ar][ac + 8 * u], asrc + 8 * u);
      stage16B(&Bts[ar][ac + 8 * u], bsrc + 8 * u);
    }
    if (k0 + 64 < K) {  // gfx1250 global_prefetch_b8 for next tiles
      __builtin_prefetch(asrc + 64, 0, 3);
      __builtin_prefetch(bsrc + 64, 0, 3);
    }
    stage_wait();
    __syncthreads();

#pragma unroll
    for (int c = 0; c < 2; ++c) {
      bf16x16 af = frag_a(&As[wave * 16 + m][0], c * 32, half);
#pragma unroll
      for (int s = 0; s < 4; ++s) {
        bf16x16 bfv = frag_b(&Bts[s * 16 + m][0], c * 32, half);
        acc[s] = __builtin_amdgcn_wmma_f32_16x16x32_bf16(
            false, af, false, bfv, (short)0, acc[s], false, false);
      }
    }
    __syncthreads();
  }

#pragma unroll
  for (int s = 0; s < 4; ++s)
#pragma unroll
    for (int r = 0; r < 8; ++r) {
      int row = bm + wave * 16 + half * 8 + r;   // C layout: VGPR r -> M=r / r+8
      int col = bn + s * 16 + m;
      C[(size_t)row * N + col] = acc[s][r] + bias[col];
    }
}

// --------------------------- RoPE (adjacent pairs) + repack; Q/K row-major
// [NH][S][D], V transposed d-major [NH][D][S] so P@V B-fragments are contiguous.
__global__ void rope_pack(const float* __restrict__ qf, const float* __restrict__ kf,
                          const float* __restrict__ vf, const int* __restrict__ pos_ids,
                          bf16_t* __restrict__ qb, bf16_t* __restrict__ kb,
                          bf16_t* __restrict__ vb) {
  int t = blockIdx.x * blockDim.x + threadIdx.x;  // (h, s, pair)
  int j = t & 31;                 // pair index within head dim (D/2 = 32)
  int s = (t >> 5) & (S_LEN - 1);
  int h = t >> 16;                // 32 * 2048 = 65536 per head
  if (h >= NHEADS) return;
  float inv = __powf(10000.0f, -(float)j * (1.0f / 32.0f));
  float ang = (float)pos_ids[s] * inv;
  float cs = __cosf(ang), sn = __sinf(ang);
  size_t gsrc = (size_t)s * HDIM + (size_t)h * DHEAD + 2 * j;
  size_t gdst = ((size_t)h * S_LEN + s) * DHEAD + 2 * j;
  float qe = qf[gsrc], qo = qf[gsrc + 1];
  float ke = kf[gsrc], ko = kf[gsrc + 1];
  qb[gdst]     = (bf16_t)(qe * cs - qo * sn);
  qb[gdst + 1] = (bf16_t)(qo * cs + qe * sn);
  kb[gdst]     = (bf16_t)(ke * cs - ko * sn);
  kb[gdst + 1] = (bf16_t)(ko * cs + ke * sn);
  size_t vdst = ((size_t)h * DHEAD + 2 * j) * S_LEN + s;   // [NH][D][S]
  vb[vdst]         = (bf16_t)vf[gsrc];
  vb[vdst + S_LEN] = (bf16_t)vf[gsrc + 1];
}

// ------------------------------------------- flash attention, WMMA everywhere
// grid (S/64, NH); block 128 = 4 waves; wave owns 16 query rows.
__global__ __launch_bounds__(128)
void attn_flash_wmma(const bf16_t* __restrict__ Qb, const bf16_t* __restrict__ Kb,
                     const bf16_t* __restrict__ Vb, const float* __restrict__ mask,
                     float* __restrict__ ctx) {
  __shared__ __attribute__((aligned(16))) bf16_t Ks[64][64];  // [key][d]
  __shared__ __attribute__((aligned(16))) bf16_t Vs[64][64];  // [d][key]
  __shared__ __attribute__((aligned(16))) bf16_t Ps[64][64];  // [qrow][key]
  const int tid = threadIdx.x, lane = tid & 31, wave = tid >> 5;
  const int m = lane & 15, half = lane >> 4;
  const int h = blockIdx.y;
  const int q0 = blockIdx.x * 64;
  const bf16_t* Qh = Qb + (size_t)h * S_LEN * DHEAD;
  const bf16_t* Kh = Kb + (size_t)h * S_LEN * DHEAD;
  const bf16_t* Vh = Vb + (size_t)h * DHEAD * S_LEN;  // d-major

  const bf16_t* qrow = &Qh[(size_t)(q0 + wave * 16 + m) * DHEAD];
  bf16x16 qfrag[2];
#pragma unroll
  for (int c = 0; c < 2; ++c) qfrag[c] = frag_a(qrow, c * 32, half);

  f32x8 accO[4] = {};
  float mi[8], li[8];
#pragma unroll
  for (int r = 0; r < 8; ++r) { mi[r] = -3.0e30f; li[r] = 0.0f; }

  const int sr = tid >> 1, scc = (tid & 1) * 32;  // staging row / col-half
  for (int kt = 0; kt < S_LEN; kt += 64) {
    __syncthreads();  // prior iteration done reading Ks/Vs
    { // stage K [key][d] and V [d][key] straight into LDS
      const bf16_t* ksrc = &Kh[(size_t)(kt + sr) * DHEAD + scc];
      const bf16_t* vsrc = &Vh[(size_t)sr * S_LEN + kt + scc];
#pragma unroll
      for (int u = 0; u < 4; ++u) {
        stage16B(&Ks[sr][scc + 8 * u], ksrc + 8 * u);
        stage16B(&Vs[sr][scc + 8 * u], vsrc + 8 * u);
      }
    }
    stage_wait();
    __syncthreads();

    // scores S = Q @ K^T (contraction over d), scaled + masked
    f32x8 sc[4];
#pragma unroll
    for (int s = 0; s < 4; ++s) {
      f32x8 z = {};
#pragma unroll
      for (int c = 0; c < 2; ++c) {
        bf16x16 kx = frag_b(&Ks[s * 16 + m][0], c * 32, half);
        z = __builtin_amdgcn_wmma_f32_16x16x32_bf16(
            false, qfrag[c], false, kx, (short)0, z, false, false);
      }
      float mterm = (1.0f - mask[kt + s * 16 + m]) * -1.0e9f;
#pragma unroll
      for (int r = 0; r < 8; ++r) z[r] = z[r] * 0.125f + mterm;  // 1/sqrt(64)
      sc[s] = z;
    }

    // online softmax (rows live in 16-lane halves; xor<16 stays in-half)
    f32x8 rmax = sc[0];
#pragma unroll
    for (int s = 1; s < 4; ++s)
#pragma unroll
      for (int r = 0; r < 8; ++r) rmax[r] = fmaxf(rmax[r], sc[s][r]);
    for (int off = 1; off < 16; off <<= 1)
#pragma unroll
      for (int r = 0; r < 8; ++r)
        rmax[r] = fmaxf(rmax[r], __shfl_xor(rmax[r], off, 32));

    float corr[8];
    f32x8 rsum = {};
#pragma unroll
    for (int r = 0; r < 8; ++r) {
      float mn = fmaxf(mi[r], rmax[r]);
      corr[r] = __expf(mi[r] - mn);
      mi[r] = mn;
    }
#pragma unroll
    for (int s = 0; s < 4; ++s)
#pragma unroll
      for (int r = 0; r < 8; ++r) {
        float p = __expf(sc[s][r] - mi[r]);
        rsum[r] += p;
        Ps[wave * 16 + half * 8 + r][s * 16 + m] = (bf16_t)p;  // C -> A reswizzle
      }
    for (int off = 1; off < 16; off <<= 1)
#pragma unroll
      for (int r = 0; r < 8; ++r) rsum[r] += __shfl_xor(rsum[r], off, 32);
#pragma unroll
    for (int r = 0; r < 8; ++r) li[r] = li[r] * corr[r] + rsum[r];
#pragma unroll
    for (int s = 0; s < 4; ++s)
#pragma unroll
      for (int r = 0; r < 8; ++r) accO[s][r] *= corr[r];

    __syncthreads();

    // O += P @ V (contraction over keys, two 32-chunks); P frag hoisted over s
#pragma unroll
    for (int c = 0; c < 2; ++c) {
      bf16x16 pf = frag_a(&Ps[wave * 16 + m][0], c * 32, half);
#pragma unroll
      for (int s = 0; s < 4; ++s) {
        bf16x16 vx = frag_b(&Vs[s * 16 + m][0], c * 32, half);  // B(k=key,n=d)
        accO[s] = __builtin_amdgcn_wmma_f32_16x16x32_bf16(
            false, pf, false, vx, (short)0, accO[s], false, false);
      }
    }
  }

  // ctx[S, H] fp32, normalized by row sums
#pragma unroll
  for (int s = 0; s < 4; ++s)
#pragma unroll
    for (int r = 0; r < 8; ++r) {
      int row = q0 + wave * 16 + half * 8 + r;
      int col = h * DHEAD + s * 16 + m;
      ctx[(size_t)row * HDIM + col] = accO[s][r] / li[r];
    }
}

// ----------------------------------------------------------------- launcher
extern "C" void kernel_launch(void* const* d_in, const int* in_sizes, int n_in,
                              void* d_out, int out_size, void* d_ws, size_t ws_size,
                              hipStream_t stream) {
  (void)in_sizes; (void)n_in; (void)out_size; (void)ws_size;
  const float* hidden = (const float*)d_in[0];
  const float* mask   = (const float*)d_in[1];
  const int*   posids = (const int*)d_in[2];
  const float* Wq = (const float*)d_in[3];
  const float* bq = (const float*)d_in[4];
  const float* Wk = (const float*)d_in[5];
  const float* bk = (const float*)d_in[6];
  const float* Wv = (const float*)d_in[7];
  const float* bv = (const float*)d_in[8];
  const float* Wo = (const float*)d_in[9];
  const float* bo = (const float*)d_in[10];
  float* out = (float*)d_out;

  char* ws = (char*)d_ws;
  const size_t MB = 1ull << 20;
  bf16_t* hb   = (bf16_t*)(ws + 0);        // 4 MB
  bf16_t* wqb  = (bf16_t*)(ws + 4 * MB);   // 2 MB each, transposed [N][K]
  bf16_t* wkb  = (bf16_t*)(ws + 6 * MB);
  bf16_t* wvb  = (bf16_t*)(ws + 8 * MB);
  bf16_t* wob  = (bf16_t*)(ws + 10 * MB);
  float*  qf   = (float*)(ws + 12 * MB);   // 8 MB each
  float*  kf   = (float*)(ws + 20 * MB);
  float*  vf   = (float*)(ws + 28 * MB);
  bf16_t* qb   = (bf16_t*)(ws + 36 * MB);  // 4 MB each
  bf16_t* kb   = (bf16_t*)(ws + 40 * MB);
  bf16_t* vb   = (bf16_t*)(ws + 44 * MB);  // d-major [NH][D][S]
  float*  ctxf = qf;                       // reuse (qf consumed by rope_pack)
  bf16_t* ctxb = (bf16_t*)(ws + 20 * MB);  // reuse kf

  const int NHID = S_LEN * HDIM;   // 2M elems
  const int NW   = HDIM * HDIM;    // 1M elems

  cvt_f32_to_bf16<<<(NHID + 255) / 256, 256, 0, stream>>>(hidden, hb, NHID);
  cvt_transpose_bf16<<<(NW + 255) / 256, 256, 0, stream>>>(Wq, wqb, HDIM, HDIM);
  cvt_transpose_bf16<<<(NW + 255) / 256, 256, 0, stream>>>(Wk, wkb, HDIM, HDIM);
  cvt_transpose_bf16<<<(NW + 255) / 256, 256, 0, stream>>>(Wv, wvb, HDIM, HDIM);
  cvt_transpose_bf16<<<(NW + 255) / 256, 256, 0, stream>>>(Wo, wob, HDIM, HDIM);

  dim3 gg(HDIM / 64, S_LEN / 64);
  gemm_bf16_wmma<<<gg, 128, 0, stream>>>(hb, wqb, bq, qf, S_LEN, HDIM, HDIM);
  gemm_bf16_wmma<<<gg, 128, 0, stream>>>(hb, wkb, bk, kf, S_LEN, HDIM, HDIM);
  gemm_bf16_wmma<<<gg, 128, 0, stream>>>(hb, wvb, bv, vf, S_LEN, HDIM, HDIM);

  rope_pack<<<(NHEADS * S_LEN * 32) / 256, 256, 0, stream>>>(qf, kf, vf, posids,
                                                             qb, kb, vb);

  attn_flash_wmma<<<dim3(S_LEN / 64, NHEADS), 128, 0, stream>>>(qb, kb, vb, mask,
                                                                ctxf);

  cvt_f32_to_bf16<<<(NHID + 255) / 256, 256, 0, stream>>>(ctxf, ctxb, NHID);
  gemm_bf16_wmma<<<gg, 128, 0, stream>>>(ctxb, wob, bo, out, S_LEN, HDIM, HDIM);
}
